// MHAttention_12180527251523
// MI455X (gfx1250) — matrix operand order repaired and checked by reading, not compiled, not run on hardware
//
#include <hip/hip_runtime.h>
#include <hip/hip_bf16.h>

typedef __bf16 bf16_t;
typedef __attribute__((ext_vector_type(16))) __bf16 v16bf;
typedef __attribute__((ext_vector_type(8)))  __bf16 v8bf;
typedef __attribute__((ext_vector_type(8)))  float   v8f;

#define D_MODEL 768
#define N_HEAD  12
#define D_HEAD  64
#define BATCH   2
#define SEQ     2048
#define MTOT    (BATCH * SEQ)
#define MASK_FILL (-100000.0f)

// ---------------------------------------------------------------------------
// WMMA fragment loaders (A/B in this code's contiguous k order; C/D per CDNA5 ISA 7.12.2, wave32)
// A (16x32 bf16, MxK): lane<16 -> row=lane,    K=k0+0..15 (contig)
//                      lane>=16 -> row=lane-16, K=k0+16..31 (contig)
// B (32x16 bf16, KxN): lane holds column N=n0+(lane&15);
//                      lane<16 -> K=k0+0..15, lane>=16 -> K=k0+16..31 (contig)
// C/D (16x16 f32): vgpr r: lanes0-15 -> M=r, lanes16-31 -> M=r+8; N=lane&15
// ---------------------------------------------------------------------------
static __device__ __forceinline__ v16bf load_fragA(const bf16_t* base, int ld,
                                                   int m0, int k0, int lane) {
  const int row = m0 + (lane & 15);
  const int kA  = k0 + ((lane < 16) ? 0 : 16);
  const int kB  = kA + 8;
  const bf16_t* p = base + (size_t)row * ld;
  v8bf lo = *(const v8bf*)(p + kA);
  v8bf hi = *(const v8bf*)(p + kB);
  v16bf a;
#pragma unroll
  for (int i = 0; i < 8; ++i) { a[i] = lo[i]; a[8 + i] = hi[i]; }
  return a;
}

// source is row-major with the WMMA K dimension contiguous (ld = row stride)
static __device__ __forceinline__ v16bf load_fragB(const bf16_t* base, int ld,
                                                   int n0, int k0, int lane) {
  const int n  = n0 + (lane & 15);
  const int kb = k0 + ((lane < 16) ? 0 : 16);
  return *(const v16bf*)(base + (size_t)n * ld + kb);
}

static __device__ __forceinline__ v8f wmma_bf16(v16bf a, v16bf b, v8f c) {
  return __builtin_amdgcn_wmma_f32_16x16x32_bf16(
      /*neg_a=*/false, a, /*neg_b=*/false, b,
      /*c_mod=*/(short)0, c, /*reuse_a=*/false, /*reuse_b=*/false);
}

// ---------------------------------------------------------------------------
// fp32 -> bf16 conversion (vectorized 4-wide)
// ---------------------------------------------------------------------------
__global__ void cvt_f32_bf16_kernel(const float* __restrict__ in,
                                    bf16_t* __restrict__ out, int n) {
  int i = (blockIdx.x * blockDim.x + threadIdx.x) * 4;
  if (i + 3 < n) {
    float4 v = *(const float4*)(in + i);
    out[i]     = (bf16_t)v.x;
    out[i + 1] = (bf16_t)v.y;
    out[i + 2] = (bf16_t)v.z;
    out[i + 3] = (bf16_t)v.w;
  } else {
    for (; i < n; ++i) out[i] = (bf16_t)in[i];
  }
}

// ---------------------------------------------------------------------------
// Y = X @ W^T + bias. X:[4096,768] bf16, W:[768,768] bf16 row-major (so W^T
// has contiguous K per output column). One wave computes a 16x64 tile: the A
// fragment is loaded once per k-step and reused across 4 B fragments / 4
// WMMAs (4x better A reuse, 1.6x less load traffic than 16x16 tiles).
// mode 0: bf16 out, layout [B,H,S,Dh]      (Q, K projections)
// mode 1: bf16 out, layout [B,H,Dh,S]      (V projection, pre-transposed)
// mode 2: f32  out, layout [M, D_MODEL]    (final output projection)
// ---------------------------------------------------------------------------
__global__ void gemm_wmma_kernel(const bf16_t* __restrict__ X,
                                 const bf16_t* __restrict__ W,
                                 const float* __restrict__ bias,
                                 void* __restrict__ out, int mode) {
  const int lane = threadIdx.x & 31;
  const int wave = threadIdx.x >> 5;
  const int m0 = blockIdx.x * 16;
  const int n0 = blockIdx.y * 256 + wave * 64;

  const v8f vzero = {};
  v8f acc[4];
#pragma unroll
  for (int f = 0; f < 4; ++f) acc[f] = vzero;

#pragma unroll 2
  for (int k0 = 0; k0 < D_MODEL; k0 += 32) {
    v16bf a = load_fragA(X, D_MODEL, m0, k0, lane);
    if (k0 + 64 < D_MODEL) {
      // prefetch the next-but-one A k-tile (lowers to global_prefetch_b8)
      __builtin_prefetch(X + (size_t)(m0 + (lane & 15)) * D_MODEL + k0 + 64, 0, 1);
    }
#pragma unroll
    for (int f = 0; f < 4; ++f) {
      v16bf b = load_fragB(W, D_MODEL, n0 + f * 16, k0, lane);
      acc[f] = wmma_bf16(a, b, acc[f]);
    }
  }

#pragma unroll
  for (int f = 0; f < 4; ++f) {
    const int n = n0 + f * 16 + (lane & 15);
    const float bn = bias[n];
#pragma unroll
    for (int r = 0; r < 8; ++r) {
      const int m = m0 + r + ((lane < 16) ? 0 : 8);
      const float v = acc[f][r] + bn;
      if (mode == 2) {
        ((float*)out)[(size_t)m * D_MODEL + n] = v;
      } else {
        const int b_ = m >> 11, s = m & (SEQ - 1);
        const int h = n >> 6, d = n & 63;
        const size_t idx =
            (mode == 0)
                ? ((((size_t)b_ * N_HEAD + h) * SEQ + s) * D_HEAD + d)
                : ((((size_t)b_ * N_HEAD + h) * D_HEAD + d) * SEQ + s);
        ((bf16_t*)out)[idx] = (bf16_t)v;
      }
    }
  }
}

// ---------------------------------------------------------------------------
// Causal flash attention. One wave owns a 16-row query tile of one (b,h).
// Key tiles of 32: scores via 4 WMMAs, online softmax (row lives across a
// 16-lane half-wave -> shfl_xor butterflies), P relayout through per-wave LDS
// (C-layout f32 -> A-layout bf16), PV via 4 WMMAs against pre-transposed V.
// Keeps the O(S^2) score tensor entirely in registers (saves ~400 MB of HBM
// round-trip vs materializing scores -> this is the dominant optimization).
// ---------------------------------------------------------------------------
__global__ void flash_attn_kernel(const bf16_t* __restrict__ Q,
                                  const bf16_t* __restrict__ Kc,
                                  const bf16_t* __restrict__ Vt,
                                  bf16_t* __restrict__ ctx) {
  __shared__ bf16_t Plds[4][16][32];  // one 16x32 P tile per wave
  const int lane = threadIdx.x & 31;
  const int wave = threadIdx.x >> 5;
  const int tile = blockIdx.x * 4 + wave;
  const int qtPerBH = SEQ / 16;  // 128
  const int bh = tile / qtPerBH;
  const int q0 = (tile % qtPerBH) * 16;
  const int b = bh / N_HEAD, h = bh % N_HEAD;

  const bf16_t* Qb = Q  + (size_t)bh * SEQ * D_HEAD;
  const bf16_t* Kb = Kc + (size_t)bh * SEQ * D_HEAD;
  const bf16_t* Vb = Vt + (size_t)bh * D_HEAD * SEQ;

  // Q tile fragments are reused for every key tile
  const v16bf qa0 = load_fragA(Qb, D_HEAD, q0, 0,  lane);
  const v16bf qa1 = load_fragA(Qb, D_HEAD, q0, 32, lane);

  const v8f vzero = {};
  v8f acc[4];
#pragma unroll
  for (int f = 0; f < 4; ++f) acc[f] = vzero;
  float mrow[8], lrow[8];
#pragma unroll
  for (int r = 0; r < 8; ++r) { mrow[r] = -1e30f; lrow[r] = 0.0f; }

  const int ktmax = (q0 + 15) >> 5;  // causal: only lower-triangle key tiles
  for (int kt = 0; kt <= ktmax; ++kt) {
    const int kb0 = kt * 32;

    // scores for 32 keys: S = (Q K^T) -- K rows have contiguous Dh, so the
    // B fragment (K-dim = Dh) is a straight 32B load per lane.
    v8f s0 = vzero, s1 = vzero;
    {
      v16bf b00 = load_fragB(Kb, D_HEAD, kb0,      0,  lane);
      v16bf b01 = load_fragB(Kb, D_HEAD, kb0,      32, lane);
      v16bf b10 = load_fragB(Kb, D_HEAD, kb0 + 16, 0,  lane);
      v16bf b11 = load_fragB(Kb, D_HEAD, kb0 + 16, 32, lane);
      s0 = wmma_bf16(qa0, b00, s0);
      s0 = wmma_bf16(qa1, b01, s0);
      s1 = wmma_bf16(qa0, b10, s1);
      s1 = wmma_bf16(qa1, b11, s1);
    }

    const int j0 = kb0 + (lane & 15);
    const int j1 = j0 + 16;
    float alpha[8];
#pragma unroll
    for (int r = 0; r < 8; ++r) {
      const int m = q0 + r + ((lane < 16) ? 0 : 8);
      float v0 = (j0 <= m) ? s0[r] * 0.125f : MASK_FILL;  // 1/sqrt(64)
      float v1 = (j1 <= m) ? s1[r] * 0.125f : MASK_FILL;
      float t = fmaxf(v0, v1);
#pragma unroll
      for (int off = 1; off < 16; off <<= 1)
        t = fmaxf(t, __shfl_xor(t, off, 32));
      const float mn = fmaxf(mrow[r], t);
      const float a  = __expf(mrow[r] - mn);  // 0 on first tile
      const float p0 = __expf(v0 - mn);
      const float p1 = __expf(v1 - mn);
      float ps = p0 + p1;
#pragma unroll
      for (int off = 1; off < 16; off <<= 1)
        ps += __shfl_xor(ps, off, 32);
      lrow[r] = lrow[r] * a + ps;
      mrow[r] = mn;
      alpha[r] = a;
      // stage P (C layout -> [row][key] in LDS, bf16)
      const int mr = r + ((lane < 16) ? 0 : 8);
      Plds[wave][mr][lane & 15]        = (bf16_t)p0;
      Plds[wave][mr][16 + (lane & 15)] = (bf16_t)p1;
    }

#pragma unroll
    for (int f = 0; f < 4; ++f)
#pragma unroll
      for (int r = 0; r < 8; ++r) acc[f][r] *= alpha[r];

    // re-read P as an A fragment (per-wave LDS, DS ops are in-order per wave)
    const v16bf pa = load_fragA(&Plds[wave][0][0], 32, 0, 0, lane);

    // PV: V stored transposed [Dh][S] -> B fragment (K-dim = key) contiguous
#pragma unroll
    for (int f = 0; f < 4; ++f) {
      v16bf vb = load_fragB(Vb, SEQ, f * 16, kb0, lane);
      acc[f] = wmma_bf16(pa, vb, acc[f]);
    }
  }

  // normalize and scatter ctx back to [B,S,D_MODEL] (bf16 for final GEMM)
#pragma unroll
  for (int r = 0; r < 8; ++r) {
    const int m = q0 + r + ((lane < 16) ? 0 : 8);
    const float inv = 1.0f / lrow[r];
#pragma unroll
    for (int f = 0; f < 4; ++f) {
      const int d = h * D_HEAD + f * 16 + (lane & 15);
      ctx[((size_t)b * SEQ + m) * D_MODEL + d] = (bf16_t)(acc[f][r] * inv);
    }
  }
}

// ---------------------------------------------------------------------------
// Launch
// ---------------------------------------------------------------------------
extern "C" void kernel_launch(void* const* d_in, const int* in_sizes, int n_in,
                              void* d_out, int out_size, void* d_ws,
                              size_t ws_size, hipStream_t stream) {
  (void)in_sizes; (void)n_in; (void)out_size; (void)ws_size;

  const float* xEnc = (const float*)d_in[0];
  const float* xDec = (const float*)d_in[1];
  // d_in[2] is the boolean causal mask; computed analytically in-kernel.
  const float* wQ = (const float*)d_in[3];
  const float* bQ = (const float*)d_in[4];
  const float* wK = (const float*)d_in[5];
  const float* bK = (const float*)d_in[6];
  const float* wV = (const float*)d_in[7];
  const float* bV = (const float*)d_in[8];
  const float* wO = (const float*)d_in[9];
  const float* bO = (const float*)d_in[10];

  char* ws = (char*)d_ws;
  size_t off = 0;
  auto salloc = [&](size_t elems) {
    bf16_t* p = (bf16_t*)(ws + off);
    off += ((elems * sizeof(bf16_t)) + 255) & ~(size_t)255;
    return p;
  };
  const size_t nX = (size_t)MTOT * D_MODEL;     // 3,145,728
  const size_t nW = (size_t)D_MODEL * D_MODEL;  // 589,824

  bf16_t* xEncB = salloc(nX);
  bf16_t* xDecB = salloc(nX);
  bf16_t* wQB   = salloc(nW);
  bf16_t* wKB   = salloc(nW);
  bf16_t* wVB   = salloc(nW);
  bf16_t* wOB   = salloc(nW);
  bf16_t* Qh    = salloc(nX);  // [B,H,S,Dh]
  bf16_t* Kh    = salloc(nX);  // [B,H,S,Dh]
  bf16_t* Vth   = salloc(nX);  // [B,H,Dh,S]
  bf16_t* ctx   = salloc(nX);  // [B,S,D]

  const int cvb = 256;
  cvt_f32_bf16_kernel<<<(unsigned)((nX / 4 + cvb - 1) / cvb), cvb, 0, stream>>>(xEnc, xEncB, (int)nX);
  cvt_f32_bf16_kernel<<<(unsigned)((nX / 4 + cvb - 1) / cvb), cvb, 0, stream>>>(xDec, xDecB, (int)nX);
  cvt_f32_bf16_kernel<<<(unsigned)((nW / 4 + cvb - 1) / cvb), cvb, 0, stream>>>(wQ, wQB, (int)nW);
  cvt_f32_bf16_kernel<<<(unsigned)((nW / 4 + cvb - 1) / cvb), cvb, 0, stream>>>(wK, wKB, (int)nW);
  cvt_f32_bf16_kernel<<<(unsigned)((nW / 4 + cvb - 1) / cvb), cvb, 0, stream>>>(wV, wVB, (int)nW);
  cvt_f32_bf16_kernel<<<(unsigned)((nW / 4 + cvb - 1) / cvb), cvb, 0, stream>>>(wO, wOB, (int)nW);

  // each wave: 16x64 tile; block = 4 waves -> 16 (M) x 256 (N) per block
  dim3 gGrid(MTOT / 16, D_MODEL / 256);  // 256 x 3
  gemm_wmma_kernel<<<gGrid, 128, 0, stream>>>(xDecB, wQB, bQ, Qh, 0);
  gemm_wmma_kernel<<<gGrid, 128, 0, stream>>>(xEncB, wKB, bK, Kh, 0);
  gemm_wmma_kernel<<<gGrid, 128, 0, stream>>>(xEncB, wVB, bV, Vth, 1);

  const int qTiles = BATCH * N_HEAD * (SEQ / 16);  // 3072
  flash_attn_kernel<<<qTiles / 4, 128, 0, stream>>>(Qh, Kh, Vth, ctx);

  gemm_wmma_kernel<<<gGrid, 128, 0, stream>>>(ctx, wOB, bO, d_out, 2);
}
